// ChannelWiseTensorSquareSelfInteraction_24902220383052
// MI455X (gfx1250) — matrix operand order repaired and checked by reading, not compiled, not run on hardware
//
#include <hip/hip_runtime.h>
#include <hip/hip_bf16.h>

typedef __bf16 bf16_t;
typedef bf16_t v16bf __attribute__((ext_vector_type(16)));
typedef float  v8f   __attribute__((ext_vector_type(8)));
typedef int    v4i   __attribute__((ext_vector_type(4)));

#define SQRT2F 1.41421356237309515f

#if __has_builtin(__builtin_amdgcn_global_load_async_to_lds_b128) && \
    __has_builtin(__builtin_amdgcn_s_wait_asynccnt)
#define HAVE_ASYNC_LDS 1
#else
#define HAVE_ASYNC_LDS 0
#endif

#if HAVE_ASYNC_LDS
#define WAIT_ASYNC(n) __builtin_amdgcn_s_wait_asynccnt(n)
#else
#define WAIT_ASYNC(n)
#endif

typedef __attribute__((address_space(1))) v4i* gv4p;
typedef __attribute__((address_space(3))) v4i* sv4p;

// ---------- helpers ----------

__device__ __forceinline__ void cp16(const void* g, void* s) {
#if HAVE_ASYNC_LDS
    __builtin_amdgcn_global_load_async_to_lds_b128((gv4p)g, (sv4p)s, 0, 0);
#else
    *(uint4*)s = *(const uint4*)g;
#endif
}

__device__ __forceinline__ void split_bf16(float f, unsigned short& hi, unsigned short& lo) {
    unsigned u = __float_as_uint(f);
    unsigned r = 0x7FFFu + ((u >> 16) & 1u);
    unsigned short h = (unsigned short)((u + r) >> 16);
    hi = h;
    float fh = __uint_as_float(((unsigned)h) << 16);
    float fl = f - fh;
    unsigned ul = __float_as_uint(fl);
    unsigned rl = 0x7FFFu + ((ul >> 16) & 1u);
    lo = (unsigned short)((ul + rl) >> 16);
}

__device__ __forceinline__ float silu_f(float x) {
    return x / (1.0f + __expf(-x));
}

// A fragment: 16(M) x 32(K) bf16, LDS layout A[m][k] (K contiguous, lda even).
__device__ __forceinline__ v16bf load_a_frag(const unsigned short* __restrict__ A,
                                             int mBase, int lda, int k0, int lane) {
    int m  = mBase + (lane & 15);
    int kb = (lane & 16) ? 8 : 0;
    const unsigned short* p = A + m * lda + k0 + kb;
    union { v16bf v; unsigned u[8]; } f;
#pragma unroll
    for (int j = 0; j < 4; ++j) f.u[j]     = *(const unsigned*)(p + 2 * j);
#pragma unroll
    for (int j = 0; j < 4; ++j) f.u[4 + j] = *(const unsigned*)(p + 16 + 2 * j);
    return f.v;
}

// B fragment: 32(K) x 16(N) bf16, LDS layout B[n][k] (k-tile = 32).
__device__ __forceinline__ v16bf load_b_frag(const unsigned short* __restrict__ B,
                                             int nBase, int lane) {
    int nn = nBase + (lane & 15);
    int kb = (lane & 16) ? 8 : 0;
    const unsigned short* p = B + nn * 32 + kb;
    union { v16bf v; unsigned u[8]; } f;
#pragma unroll
    for (int j = 0; j < 8; ++j) f.u[j] = *(const unsigned*)(p + 2 * j + ((j >> 2) << 3));
    return f.v;
}

__device__ __forceinline__ v8f wmma3(v16bf ah, v16bf al, v16bf bh, v16bf bl, v8f c) {
    c = __builtin_amdgcn_wmma_f32_16x16x32_bf16(false, ah, false, bh, (short)0, c, false, false);
    c = __builtin_amdgcn_wmma_f32_16x16x32_bf16(false, al, false, bh, (short)0, c, false, false);
    c = __builtin_amdgcn_wmma_f32_16x16x32_bf16(false, ah, false, bl, (short)0, c, false, false);
    return c;
}

// Stage one 128(N) x 32(K) hi/lo B tile from pre-transposed planes [N][Ktot].
// 16 KB total = 1024 16-byte chunks; 8 waves x 4 instrs x 32 lanes.
__device__ __forceinline__ void issue_b_tile(const unsigned short* __restrict__ wh,
                                             const unsigned short* __restrict__ wl,
                                             int Ktot, int ncol0, int k0,
                                             unsigned short* Bh, unsigned short* Bl,
                                             int wave, int lane) {
#pragma unroll
    for (int c = 0; c < 4; ++c) {
        int g = wave * 128 + c * 32 + lane;
        int within = g & 511;
        int row = within >> 2;
        int off = (within & 3) * 8;  // ushort offset (16B chunks)
        const unsigned short* src = ((g >> 9) ? wl : wh) + (size_t)(ncol0 + row) * Ktot + k0 + off;
        unsigned short* dst = ((g >> 9) ? Bl : Bh) + row * 32 + off;
        cp16(src, dst);
    }
}

// ---------- kernel 0: weight -> transposed hi/lo bf16 planes ----------
// dst[n][k] planes from src[k][n] (row stride = `stride`), K = reduction dim.

__global__ __launch_bounds__(256) void k0_convert(const float* __restrict__ src, int K, int N,
                                                  int stride,
                                                  unsigned short* __restrict__ dh,
                                                  unsigned short* __restrict__ dl) {
    int idx = blockIdx.x * 256 + threadIdx.x;
    if (idx >= K * N) return;
    int nn = idx / K, k = idx - nn * K;
    unsigned short hi, lo;
    split_bf16(src[(size_t)k * stride + nn], hi, lo);
    dh[(size_t)nn * K + k] = hi;
    dl[(size_t)nn * K + k] = lo;
}

// ---------- kernel 1: scal = [s, s^2, |v|^2]; h1 = silu(scal @ W1) -> hi/lo planes ----------
// block = 256 threads (8 waves); tile M=64 rows, N=128 cols (gridDim.y = 3), K=384.

__global__ __launch_bounds__(256) void k1_feat_gemm1(const float* __restrict__ x,
                                                     const unsigned short* __restrict__ w1h,
                                                     const unsigned short* __restrict__ w1l,
                                                     unsigned short* __restrict__ h1h,
                                                     unsigned short* __restrict__ h1l, int n) {
    __shared__ __align__(16) unsigned short Ah[64 * 384];
    __shared__ __align__(16) unsigned short Al[64 * 384];
    __shared__ __align__(16) unsigned short Bh[2][128 * 32];
    __shared__ __align__(16) unsigned short Bl[2][128 * 32];

    const int tid = threadIdx.x;
    const int lane = tid & 31, wave = tid >> 5;
    const int row0 = blockIdx.x * 64;
    const int ncol0 = blockIdx.y * 128;

    issue_b_tile(w1h, w1l, 384, ncol0, 0, Bh[0], Bl[0], wave, lane);

    // featurize 64x384 scal tile into hi/lo bf16
    for (int idx = tid; idx < 64 * 384; idx += 256) {
        int rl = idx / 384, c = idx - rl * 384;
        int r = row0 + rl;
        float f = 0.f;
        if (r < n) {
            const float* xr = x + (size_t)r * 512;
            if (c < 128) {
                f = xr[c];
            } else if (c < 256) {
                float s = xr[c - 128];
                f = s * s;
            } else {
                int m = c - 256;
                float a = xr[128 + 3 * m], b = xr[129 + 3 * m], d = xr[130 + 3 * m];
                f = a * a + b * b + d * d;
            }
        }
        unsigned short hi, lo;
        split_bf16(f, hi, lo);
        Ah[rl * 384 + c] = hi;
        Al[rl * 384 + c] = lo;
    }

    const int mt = wave & 3;   // row tile (16 rows)
    const int nh = wave >> 2;  // column half (64 cols = 4 N-tiles)
    v8f acc[4] = {};

    for (int kt = 0; kt < 12; ++kt) {
        if (kt + 1 < 12) {
            issue_b_tile(w1h, w1l, 384, ncol0, (kt + 1) * 32, Bh[(kt + 1) & 1], Bl[(kt + 1) & 1],
                         wave, lane);
            WAIT_ASYNC(4);
        } else {
            WAIT_ASYNC(0);
        }
        __syncthreads();
        const unsigned short* bhp = Bh[kt & 1];
        const unsigned short* blp = Bl[kt & 1];
        v16bf ah = load_a_frag(Ah, 16 * mt, 384, kt * 32, lane);
        v16bf al = load_a_frag(Al, 16 * mt, 384, kt * 32, lane);
#pragma unroll
        for (int t = 0; t < 4; ++t) {
            v16bf bh = load_b_frag(bhp, nh * 64 + t * 16, lane);
            v16bf bl = load_b_frag(blp, nh * 64 + t * 16, lane);
            acc[t] = wmma3(ah, al, bh, bl, acc[t]);
        }
        __syncthreads();
    }

    const int half = lane >> 4;
#pragma unroll
    for (int t = 0; t < 4; ++t) {
#pragma unroll
        for (int r = 0; r < 8; ++r) {
            int row = row0 + 16 * mt + r + 8 * half;
            int col = ncol0 + nh * 64 + t * 16 + (lane & 15);
            if (row < n) {
                unsigned short hi, lo;
                split_bf16(silu_f(acc[t][r]), hi, lo);
                h1h[(size_t)row * 384 + col] = hi;
                h1l[(size_t)row * 384 + col] = lo;
            }
        }
    }
}

// ---------- kernel 2: gates = silu(h1 @ W2[:, :640]); gated linears; norms ----------
// block = 256 threads (8 waves); tile M=32 rows.

__global__ __launch_bounds__(256) void k2_gates_out(
    const float* __restrict__ x, const unsigned short* __restrict__ h1h,
    const unsigned short* __restrict__ h1l, const unsigned short* __restrict__ w2h,
    const unsigned short* __restrict__ w2l, const unsigned short* __restrict__ lwsh,
    const unsigned short* __restrict__ lwsl, const unsigned short* __restrict__ lwvh,
    const unsigned short* __restrict__ lwvl, float* __restrict__ out, int n) {
    __shared__ float gates[32 * 640];                      // 80 KB
    __shared__ __align__(16) unsigned short Fh[32 * 384];  // 24 KB (h1 -> a -> b_i)
    __shared__ __align__(16) unsigned short Fl[32 * 384];  // 24 KB
    __shared__ __align__(16) unsigned short Bh[2][128 * 32];  // 16 KB
    __shared__ __align__(16) unsigned short Bl[2][128 * 32];  // 16 KB
    // reduction scratch aliased on gates[] (zeroed only after gates are dead)
    float* redS = gates;
    float* redQ = gates + 32;
    float* redV = gates + 64;

    const int tid = threadIdx.x;
    const int lane = tid & 31, wave = tid >> 5;
    const int half = lane >> 4;
    const int row0 = blockIdx.x * 32;
    const int mt = wave & 1;   // row tile (16 rows)
    const int nq = wave >> 1;  // column quarter (32 cols = 2 N-tiles)

    // ---- stage h1 tile (32x384, hi/lo planes) ----
    if (row0 + 32 <= n) {
        // contiguous 24 KB per plane -> 3072 16B chunks, 12 per thread (plane-uniform per step)
#pragma unroll
        for (int c = 0; c < 12; ++c) {
            int g = c * 256 + tid;
            const unsigned short* srcp =
                (g < 1536) ? (h1h + (size_t)row0 * 384) : (h1l + (size_t)row0 * 384);
            unsigned short* dstp = (g < 1536) ? Fh : Fl;
            int wi = (g < 1536) ? g : g - 1536;
            cp16(srcp + wi * 8, dstp + wi * 8);
        }
    } else {
        for (int idx = tid; idx < 32 * 384; idx += 256) {
            int rl = idx / 384, c = idx - rl * 384;
            int r = row0 + rl;
            Fh[idx] = (r < n) ? h1h[(size_t)r * 384 + c] : (unsigned short)0;
            Fl[idx] = (r < n) ? h1l[(size_t)r * 384 + c] : (unsigned short)0;
        }
    }

    // ---- gates GEMM: (32x384) @ W2[:, :640], SiLU, 5 column chunks of 128 ----
    for (int nc = 0; nc < 5; ++nc) {
        v8f acc[2] = {};
        issue_b_tile(w2h, w2l, 384, nc * 128, 0, Bh[0], Bl[0], wave, lane);
        for (int kt = 0; kt < 12; ++kt) {
            if (kt + 1 < 12) {
                issue_b_tile(w2h, w2l, 384, nc * 128, (kt + 1) * 32, Bh[(kt + 1) & 1],
                             Bl[(kt + 1) & 1], wave, lane);
                WAIT_ASYNC(4);
            } else {
                WAIT_ASYNC(0);
            }
            __syncthreads();
            v16bf ah = load_a_frag(Fh, 16 * mt, 384, kt * 32, lane);
            v16bf al = load_a_frag(Fl, 16 * mt, 384, kt * 32, lane);
#pragma unroll
            for (int t = 0; t < 2; ++t) {
                v16bf bh = load_b_frag(Bh[kt & 1], nq * 32 + t * 16, lane);
                v16bf bl = load_b_frag(Bl[kt & 1], nq * 32 + t * 16, lane);
                acc[t] = wmma3(ah, al, bh, bl, acc[t]);
            }
            __syncthreads();
        }
#pragma unroll
        for (int t = 0; t < 2; ++t)
#pragma unroll
            for (int r = 0; r < 8; ++r) {
                int rl = 16 * mt + r + 8 * half;
                int col = nc * 128 + nq * 32 + t * 16 + (lane & 15);
                gates[rl * 640 + col] = silu_f(acc[t][r]);
            }
    }
    __syncthreads();

    // ---- build a = scal * gs (reuse Fh/Fl) ----
    for (int idx = tid; idx < 32 * 384; idx += 256) {
        int rl = idx / 384, c = idx - rl * 384;
        int r = row0 + rl;
        float sc = 0.f;
        if (r < n) {
            const float* xr = x + (size_t)r * 512;
            if (c < 128) {
                sc = xr[c];
            } else if (c < 256) {
                float s = xr[c - 128];
                sc = s * s;
            } else {
                int m = c - 256;
                float a = xr[128 + 3 * m], b = xr[129 + 3 * m], d = xr[130 + 3 * m];
                sc = a * a + b * b + d * d;
            }
        }
        float f = sc * gates[rl * 640 + c];
        unsigned short hi, lo;
        split_bf16(f, hi, lo);
        Fh[rl * 384 + c] = hi;
        Fl[rl * 384 + c] = lo;
    }

    // ---- out_s = a @ lin_ws (K=384) ----
    v8f accS[2] = {};
    issue_b_tile(lwsh, lwsl, 384, 0, 0, Bh[0], Bl[0], wave, lane);
    for (int kt = 0; kt < 12; ++kt) {
        if (kt + 1 < 12) {
            issue_b_tile(lwsh, lwsl, 384, 0, (kt + 1) * 32, Bh[(kt + 1) & 1], Bl[(kt + 1) & 1],
                         wave, lane);
            WAIT_ASYNC(4);
        } else {
            WAIT_ASYNC(0);
        }
        __syncthreads();
        v16bf ah = load_a_frag(Fh, 16 * mt, 384, kt * 32, lane);
        v16bf al = load_a_frag(Fl, 16 * mt, 384, kt * 32, lane);
#pragma unroll
        for (int t = 0; t < 2; ++t) {
            v16bf bh = load_b_frag(Bh[kt & 1], nq * 32 + t * 16, lane);
            v16bf bl = load_b_frag(Bl[kt & 1], nq * 32 + t * 16, lane);
            accS[t] = wmma3(ah, al, bh, bl, accS[t]);
        }
        __syncthreads();
    }

    // ---- out_v_i = (vecs_i * gv) @ lin_wv (K=256), i = 0..2 ----
    v8f accV[3][2] = {};
    for (int i = 0; i < 3; ++i) {
        __syncthreads();
        for (int idx = tid; idx < 32 * 256; idx += 256) {
            int rl = idx >> 8, u = idx & 255;
            int r = row0 + rl;
            float f = 0.f;
            if (r < n) {
                const float* xr = x + (size_t)r * 512;
                float g = gates[rl * 640 + 384 + u];
                if (u < 128) {
                    f = xr[128 + 3 * u + i] * g;
                } else {
                    int m = u - 128;
                    f = SQRT2F * xr[m] * xr[128 + 3 * m + i] * g;
                }
            }
            unsigned short hi, lo;
            split_bf16(f, hi, lo);
            Fh[rl * 256 + u] = hi;
            Fl[rl * 256 + u] = lo;
        }
        issue_b_tile(lwvh, lwvl, 256, 0, 0, Bh[0], Bl[0], wave, lane);
        for (int kt = 0; kt < 8; ++kt) {
            if (kt + 1 < 8) {
                issue_b_tile(lwvh, lwvl, 256, 0, (kt + 1) * 32, Bh[(kt + 1) & 1], Bl[(kt + 1) & 1],
                             wave, lane);
                WAIT_ASYNC(4);
            } else {
                WAIT_ASYNC(0);
            }
            __syncthreads();
            v16bf ah = load_a_frag(Fh, 16 * mt, 256, kt * 32, lane);
            v16bf al = load_a_frag(Fl, 16 * mt, 256, kt * 32, lane);
#pragma unroll
            for (int t = 0; t < 2; ++t) {
                v16bf bh = load_b_frag(Bh[kt & 1], nq * 32 + t * 16, lane);
                v16bf bl = load_b_frag(Bl[kt & 1], nq * 32 + t * 16, lane);
                accV[i][t] = wmma3(ah, al, bh, bl, accV[i][t]);
            }
            __syncthreads();
        }
    }
    __syncthreads();
    if (tid < 32) { redS[tid] = 0.f; redQ[tid] = 0.f; redV[tid] = 0.f; }
    __syncthreads();

    // ---- epilogue: residual, mean-center + RMS (out_s), vector RMS (out_v) ----
    const int colA = nq * 32 + (lane & 15);  // tile t adds +16*t
#pragma unroll
    for (int r = 0; r < 8; ++r) {
        int rl = 16 * mt + 8 * half + r;
        int rg = row0 + rl;
        const float* xr = x + (size_t)rg * 512;
        bool ok = rg < n;
#pragma unroll
        for (int t = 0; t < 2; ++t) {
            int col = colA + 16 * t;
            accS[t][r] += ok ? xr[col] : 0.f;
#pragma unroll
            for (int i = 0; i < 3; ++i) accV[i][t][r] += ok ? xr[128 + 3 * col + i] : 0.f;
        }
    }

    // round 1: row sums of out_s and |out_v|^2
#pragma unroll
    for (int r = 0; r < 8; ++r) {
        float ps = accS[0][r] + accS[1][r];
        float pv = 0.f;
#pragma unroll
        for (int t = 0; t < 2; ++t)
#pragma unroll
            for (int i = 0; i < 3; ++i) pv += accV[i][t][r] * accV[i][t][r];
#pragma unroll
        for (int m = 1; m < 16; m <<= 1) {
            ps += __shfl_xor(ps, m, 32);
            pv += __shfl_xor(pv, m, 32);
        }
        if ((lane & 15) == 0) {
            int rl = 16 * mt + 8 * half + r;
            atomicAdd(&redS[rl], ps);
            atomicAdd(&redV[rl], pv);
        }
    }
    __syncthreads();

    // center out_s, round 2: sum of squares
#pragma unroll
    for (int r = 0; r < 8; ++r) {
        int rl = 16 * mt + 8 * half + r;
        float mean = redS[rl] * (1.0f / 128.0f);
        accS[0][r] -= mean;
        accS[1][r] -= mean;
        float pq = accS[0][r] * accS[0][r] + accS[1][r] * accS[1][r];
#pragma unroll
        for (int m = 1; m < 16; m <<= 1) pq += __shfl_xor(pq, m, 32);
        if ((lane & 15) == 0) atomicAdd(&redQ[rl], pq);
    }
    __syncthreads();

    // normalize + store
#pragma unroll
    for (int r = 0; r < 8; ++r) {
        int rl = 16 * mt + 8 * half + r;
        int rg = row0 + rl;
        if (rg >= n) continue;
        float inv_s = 1.0f / (sqrtf(redQ[rl] * (1.0f / 128.0f)) + 1e-6f);
        float inv_v = 1.0f / (sqrtf(redV[rl] * (1.0f / 128.0f)) + 1e-6f);
        float* orow = out + (size_t)rg * 512;
#pragma unroll
        for (int t = 0; t < 2; ++t) {
            int col = colA + 16 * t;
            orow[col] = accS[t][r] * inv_s;
#pragma unroll
            for (int i = 0; i < 3; ++i) orow[128 + 3 * col + i] = accV[i][t][r] * inv_v;
        }
    }
}

// ---------- launch ----------

extern "C" void kernel_launch(void* const* d_in, const int* in_sizes, int n_in,
                              void* d_out, int out_size, void* d_ws, size_t ws_size,
                              hipStream_t stream) {
    const float* x   = (const float*)d_in[0];
    const float* w1  = (const float*)d_in[1];
    const float* w2  = (const float*)d_in[2];
    const float* lws = (const float*)d_in[3];
    const float* lwv = (const float*)d_in[4];
    float* out = (float*)d_out;

    int n = in_sizes[0] / 512;

    // workspace layout (all 16B-aligned)
    size_t h1e = (size_t)n * 384;
    unsigned short* h1h = (unsigned short*)d_ws;
    unsigned short* h1l = h1h + h1e;
    unsigned short* p   = h1l + h1e;
    unsigned short* w1h = p;  p += 384 * 384;
    unsigned short* w1l = p;  p += 384 * 384;
    unsigned short* w2h = p;  p += 640 * 384;
    unsigned short* w2l = p;  p += 640 * 384;
    unsigned short* lwsh = p; p += 128 * 384;
    unsigned short* lwsl = p; p += 128 * 384;
    unsigned short* lwvh = p; p += 128 * 256;
    unsigned short* lwvl = p; p += 128 * 256;

    dim3 blk(256);
    k0_convert<<<(384 * 384 + 255) / 256, blk, 0, stream>>>(w1, 384, 384, 384, w1h, w1l);
    k0_convert<<<(640 * 384 + 255) / 256, blk, 0, stream>>>(w2, 384, 640, 768, w2h, w2l);
    k0_convert<<<(128 * 384 + 255) / 256, blk, 0, stream>>>(lws, 384, 128, 128, lwsh, lwsl);
    k0_convert<<<(128 * 256 + 255) / 256, blk, 0, stream>>>(lwv, 256, 128, 128, lwvh, lwvl);

    dim3 g1((n + 63) / 64, 3);
    k1_feat_gemm1<<<g1, blk, 0, stream>>>(x, w1h, w1l, h1h, h1l, n);

    dim3 g2((n + 31) / 32);
    k2_gates_out<<<g2, blk, 0, stream>>>(x, h1h, h1l, w2h, w2l, lwsh, lwsl, lwvh, lwvl, out, n);
}
